// GCN_7078106104031
// MI455X (gfx1250) — matrix operand order repaired and checked by reading, not compiled, not run on hardware
//
#include <hip/hip_runtime.h>
#include <hip/hip_bf16.h>

typedef __attribute__((ext_vector_type(8)))  __bf16 v8bf;
typedef __attribute__((ext_vector_type(16))) __bf16 v16bf;
typedef __attribute__((ext_vector_type(8)))  float  v8f;

#define GCN_N    50000
#define GCN_DIN  512
#define GCN_H    128
#define GCN_DOUT 8
#define GCN_EPS  1e-5f

// ---------------------------------------------------------------------------
// Degree / normalization
// ---------------------------------------------------------------------------
__global__ void k_init_dinv(float* __restrict__ dinv, int n) {
    int i = blockIdx.x * blockDim.x + threadIdx.x;
    if (i < n) dinv[i] = 1.0f;              // self-loop contributes 1 to degree
}

__global__ void k_deg_accum(const int* __restrict__ dst, float* __restrict__ dinv, int e) {
    int i = blockIdx.x * blockDim.x + threadIdx.x;
    if (i < e) atomicAdd(&dinv[dst[i]], 1.0f);
}

__global__ void k_finish_dinv(float* __restrict__ dinv, int n) {
    int i = blockIdx.x * blockDim.x + threadIdx.x;
    if (i < n) dinv[i] = rsqrtf(dinv[i]);   // deg >= 1 always
}

// ---------------------------------------------------------------------------
// One-off weight prep: B[K,Nc] f32 -> bf16 fragments in per-lane WMMA order.
// Element t maps to (ct, ks, lane, i):
//   col = ct*16 + (lane&15),  k = ks*32 + (i>>3)*16 + (lane>>4)*8 + (i&7)
// so the GEMM loads one contiguous 32B v16bf per (ct, ks, lane).
// Columns beyond Nc are zero-padded (used to pad W3 from 8 to 16 cols).
// ---------------------------------------------------------------------------
__global__ void k_prep_bfrag(const float* __restrict__ B, __bf16* __restrict__ out,
                             int K, int Nc, int NCT) {
    int t = blockIdx.x * blockDim.x + threadIdx.x;
    int kSteps = K >> 5;
    int total = NCT * kSteps * 512;
    if (t >= total) return;
    int i    = t & 15;
    int lane = (t >> 4) & 31;
    int ks   = (t >> 9) % kSteps;
    int ct   = t / (kSteps << 9);
    int col  = ct * 16 + (lane & 15);
    int k    = (ks << 5) + ((i >> 3) << 4) + ((lane >> 4) << 3) + (i & 7);
    out[t] = (col < Nc) ? (__bf16)B[(size_t)k * Nc + col] : (__bf16)0.0f;
}

// ---------------------------------------------------------------------------
// WMMA GEMM: one wave -> 16 rows x (NCT*16) cols. blockDim 256 = 8 row tiles.
// A: f32 (converted in-loop) or bf16, row-major, stride K. B: prepped frags.
// C: f32 row-major, stride Nc (stores guarded by col < Nc).
// ---------------------------------------------------------------------------
template<int NCT, bool ABF16>
__global__ __launch_bounds__(256)
void k_gemm(const void* __restrict__ Av, const __bf16* __restrict__ Bfrag,
            float* __restrict__ C, int M, int K, int Nc) {
    const int lane    = threadIdx.x & 31;
    const int wave    = threadIdx.x >> 5;
    const int rowTile = blockIdx.x * 8 + wave;
    if (rowTile * 16 >= M) return;          // wave-uniform exit (EXEC stays full)

    const int l15  = lane & 15;
    const int half = lane >> 4;             // 0 or 1
    const int row  = rowTile * 16 + l15;    // A row this lane supplies
    const int kbA  = half << 3;             // A K-half base: 0 or 8
    const int kSteps = K >> 5;

    v8f acc[NCT];
#pragma unroll
    for (int ct = 0; ct < NCT; ++ct) acc[ct] = (v8f){};

    for (int ks = 0; ks < kSteps; ++ks) {
        const int k0 = ks << 5;
        v16bf af;
        if constexpr (ABF16) {
            const __bf16* ap = (const __bf16*)Av + (size_t)row * K + k0 + kbA;
            v8bf lo = *(const v8bf*)(ap);
            v8bf hi = *(const v8bf*)(ap + 16);
            af = __builtin_shufflevector(lo, hi, 0,1,2,3,4,5,6,7,8,9,10,11,12,13,14,15);
        } else {
            const float* ap = (const float*)Av + (size_t)row * K + k0 + kbA;
            float4 a0 = *(const float4*)(ap + 0);
            float4 a1 = *(const float4*)(ap + 4);
            float4 a2 = *(const float4*)(ap + 16);
            float4 a3 = *(const float4*)(ap + 20);
            af[0]=(__bf16)a0.x;  af[1]=(__bf16)a0.y;  af[2]=(__bf16)a0.z;  af[3]=(__bf16)a0.w;
            af[4]=(__bf16)a1.x;  af[5]=(__bf16)a1.y;  af[6]=(__bf16)a1.z;  af[7]=(__bf16)a1.w;
            af[8]=(__bf16)a2.x;  af[9]=(__bf16)a2.y;  af[10]=(__bf16)a2.z; af[11]=(__bf16)a2.w;
            af[12]=(__bf16)a3.x; af[13]=(__bf16)a3.y; af[14]=(__bf16)a3.z; af[15]=(__bf16)a3.w;
        }
#pragma unroll
        for (int ct = 0; ct < NCT; ++ct) {
            const v16bf bv = *(const v16bf*)(Bfrag + ((((size_t)ct * kSteps + ks) * 32 + lane) << 4));
            acc[ct] = __builtin_amdgcn_wmma_f32_16x16x32_bf16(
                          false, af, false, bv, (short)0, acc[ct], false, false);
        }
    }

    const int rbase = rowTile * 16 + (half << 3);   // D: VGPR r -> M = rbase + r
#pragma unroll
    for (int ct = 0; ct < NCT; ++ct) {
        const int col = ct * 16 + l15;
        if (col < Nc) {
#pragma unroll
            for (int r = 0; r < 8; ++r)
                C[(size_t)(rbase + r) * Nc + col] = acc[ct][r];
        }
    }
}

// ---------------------------------------------------------------------------
// Aggregation: out[i,f] = bias[f] + h[i,f]*dinv[i]^2 (self loop), then edges.
// float4-vectorized: Fq = F/4 quads per node.
// ---------------------------------------------------------------------------
__global__ void k_self_init4(const float4* __restrict__ h4, const float4* __restrict__ bias4,
                             const float* __restrict__ dinv, float4* __restrict__ out4,
                             int n, int Fq) {
    int t = blockIdx.x * blockDim.x + threadIdx.x;
    if (t >= n * Fq) return;
    int i = t / Fq, q = t - i * Fq;
    float di = dinv[i];
    float s = di * di;
    float4 hv = h4[t], bv = bias4[q];
    float4 o;
    o.x = bv.x + hv.x * s;  o.y = bv.y + hv.y * s;
    o.z = bv.z + hv.z * s;  o.w = bv.w + hv.w * s;
    out4[t] = o;
}

// wave per edge, F == 128: lane gathers float4, 4 contiguous f32 atomics
__global__ __launch_bounds__(256)
void k_edge_scatter128(const int* __restrict__ src, const int* __restrict__ dst,
                       const float* __restrict__ dinv, const float* __restrict__ h,
                       float* __restrict__ out, int e) {
    int edge = (blockIdx.x * blockDim.x + threadIdx.x) >> 5;
    if (edge >= e) return;                  // wave-uniform
    int lane = threadIdx.x & 31;
    int s = src[edge], d = dst[edge];
    float nrm = dinv[s] * dinv[d];
    const float* hs = h + (size_t)s * GCN_H + lane * 4;
    float* od = out + (size_t)d * GCN_H + lane * 4;
    float4 v = *(const float4*)hs;
    atomicAdd(od + 0, v.x * nrm);
    atomicAdd(od + 1, v.y * nrm);
    atomicAdd(od + 2, v.z * nrm);
    atomicAdd(od + 3, v.w * nrm);
}

// thread per (edge,f), F == 8
__global__ void k_edge_scatter8(const int* __restrict__ src, const int* __restrict__ dst,
                                const float* __restrict__ dinv, const float* __restrict__ h,
                                float* __restrict__ out, int e) {
    long long t = (long long)blockIdx.x * blockDim.x + threadIdx.x;
    if (t >= (long long)e * GCN_DOUT) return;
    int edge = (int)(t >> 3), f = (int)(t & 7);
    int s = src[edge], d = dst[edge];
    float nrm = dinv[s] * dinv[d];
    atomicAdd(&out[(size_t)d * GCN_DOUT + f], h[(size_t)s * GCN_DOUT + f] * nrm);
}

// ---------------------------------------------------------------------------
// BatchNorm (batch statistics) + ReLU, emitting bf16 for the next WMMA GEMM
// ---------------------------------------------------------------------------
__global__ void k_zero_stats(float* __restrict__ stats) {
    stats[threadIdx.x] = 0.0f;              // 256 threads: sum[128] + sumsq[128]
}

// 32768 threads: f = tid&127, slot = tid>>7; coalesced column partials
__global__ __launch_bounds__(256)
void k_bn_stats(const float* __restrict__ x, float* __restrict__ sum,
                float* __restrict__ sumsq, int n) {
    int tid  = blockIdx.x * blockDim.x + threadIdx.x;
    int f    = tid & (GCN_H - 1);
    int slot = tid >> 7;
    int nslots = (gridDim.x * blockDim.x) >> 7;
    float s = 0.0f, q = 0.0f;
    for (int r = slot; r < n; r += nslots) {
        float v = x[(size_t)r * GCN_H + f];
        s += v; q += v * v;
    }
    atomicAdd(&sum[f], s);
    atomicAdd(&sumsq[f], q);
}

__global__ void k_bn_finalize(const float* __restrict__ sum, const float* __restrict__ sumsq,
                              const float* __restrict__ g, const float* __restrict__ beta,
                              float* __restrict__ a, float* __restrict__ c, int n) {
    int f = threadIdx.x;
    if (f >= GCN_H) return;
    float inv_n = 1.0f / (float)n;
    float mu  = sum[f] * inv_n;
    float var = sumsq[f] * inv_n - mu * mu;   // biased variance, as reference
    float s   = g[f] * rsqrtf(var + GCN_EPS);
    a[f] = s;
    c[f] = beta[f] - mu * s;
}

__global__ void k_bn_apply_relu_bf16(const float* __restrict__ x, const float* __restrict__ a,
                                     const float* __restrict__ c, __bf16* __restrict__ y, int n) {
    int t = blockIdx.x * blockDim.x + threadIdx.x;
    if (t >= n * GCN_H) return;
    int f = t & (GCN_H - 1);
    float v = fmaxf(x[t] * a[f] + c[f], 0.0f);
    y[t] = (__bf16)v;
}

// ---------------------------------------------------------------------------
// log_softmax over 8 classes, one thread per node
// ---------------------------------------------------------------------------
__global__ void k_log_softmax8(const float* __restrict__ in, float* __restrict__ out, int n) {
    int i = blockIdx.x * blockDim.x + threadIdx.x;
    if (i >= n) return;
    const float* v = in + (size_t)i * GCN_DOUT;
    float m = v[0];
#pragma unroll
    for (int j = 1; j < GCN_DOUT; ++j) m = fmaxf(m, v[j]);
    float s = 0.0f;
#pragma unroll
    for (int j = 0; j < GCN_DOUT; ++j) s += expf(v[j] - m);
    float l = logf(s);
    float* o = out + (size_t)i * GCN_DOUT;
#pragma unroll
    for (int j = 0; j < GCN_DOUT; ++j) o[j] = v[j] - m - l;
}

// ---------------------------------------------------------------------------
// Host-side launcher
// ---------------------------------------------------------------------------
static inline int cdiv(long long a, long long b) { return (int)((a + b - 1) / b); }

extern "C" void kernel_launch(void* const* d_in, const int* in_sizes, int n_in,
                              void* d_out, int out_size, void* d_ws, size_t ws_size,
                              hipStream_t stream) {
    const float* x     = (const float*)d_in[0];
    const int*   ei    = (const int*)d_in[1];
    const float* W1    = (const float*)d_in[2];
    const float* b1    = (const float*)d_in[3];
    const float* W2    = (const float*)d_in[4];
    const float* b2    = (const float*)d_in[5];
    const float* W3    = (const float*)d_in[6];
    const float* b3    = (const float*)d_in[7];
    const float* g1    = (const float*)d_in[8];
    const float* beta1 = (const float*)d_in[9];
    const float* g2    = (const float*)d_in[10];
    const float* beta2 = (const float*)d_in[11];

    const int n = in_sizes[0] / GCN_DIN;   // 50000
    const int e = in_sizes[1] / 2;         // 1600000
    const int* src = ei;
    const int* dst = ei + e;

    // workspace carve-out (256B aligned)
    char* ws = (char*)d_ws;
    size_t off = 0;
    auto carve = [&](size_t bytes) -> void* {
        void* p = (void*)(ws + off);
        off = (off + bytes + 255) & ~(size_t)255;
        return p;
    };
    float*  dinv  = (float*)carve((size_t)n * 4);                 // rsqrt(degree)
    float*  stats = (float*)carve(4 * GCN_H * 4);                 // sum|sumsq|a|c
    float*  bufA  = (float*)carve((size_t)n * GCN_H * 4);         // GEMM out (h1/h2)
    float*  bufB  = (float*)carve((size_t)n * GCN_H * 4);         // aggregated (f32)
    __bf16* bufBF = (__bf16*)carve((size_t)n * GCN_H * 2);        // BN+ReLU out (bf16)
    float*  bufC  = (float*)carve((size_t)n * GCN_DOUT * 4);      // h3
    float*  bufD  = (float*)carve((size_t)n * GCN_DOUT * 4);      // agg3
    __bf16* W1f   = (__bf16*)carve((size_t)GCN_DIN * GCN_H * 2);  // prepped frags
    __bf16* W2f   = (__bf16*)carve((size_t)GCN_H * GCN_H * 2);
    __bf16* W3f   = (__bf16*)carve((size_t)GCN_H * 16 * 2);
    float* bn_sum = stats, *bn_sq = stats + GCN_H;
    float* bn_a = stats + 2 * GCN_H, *bn_c = stats + 3 * GCN_H;

    // --- degree / dinv + weight fragment prep ---
    k_init_dinv<<<cdiv(n, 256), 256, 0, stream>>>(dinv, n);
    k_deg_accum<<<cdiv(e, 256), 256, 0, stream>>>(dst, dinv, e);
    k_finish_dinv<<<cdiv(n, 256), 256, 0, stream>>>(dinv, n);
    k_prep_bfrag<<<cdiv((long long)GCN_DIN * GCN_H, 256), 256, 0, stream>>>(W1, W1f, GCN_DIN, GCN_H, 8);
    k_prep_bfrag<<<cdiv((long long)GCN_H * GCN_H, 256), 256, 0, stream>>>(W2, W2f, GCN_H, GCN_H, 8);
    k_prep_bfrag<<<cdiv((long long)GCN_H * 16, 256), 256, 0, stream>>>(W3, W3f, GCN_H, GCN_DOUT, 1);

    const int gemmGrid = cdiv(n, 128);     // 8 row tiles (waves) per block
    const long long nH4 = (long long)n * (GCN_H / 4);
    const long long nD4 = (long long)n * (GCN_DOUT / 4);

    // --- layer 1: GEMM (N x 512 x 128), aggregate, BN+ReLU(->bf16) ---
    k_gemm<8, false><<<gemmGrid, 256, 0, stream>>>(x, W1f, bufA, n, GCN_DIN, GCN_H);
    k_self_init4<<<cdiv(nH4, 256), 256, 0, stream>>>((const float4*)bufA, (const float4*)b1, dinv,
                                                     (float4*)bufB, n, GCN_H / 4);
    k_edge_scatter128<<<cdiv(e, 8), 256, 0, stream>>>(src, dst, dinv, bufA, bufB, e);
    k_zero_stats<<<1, 256, 0, stream>>>(stats);
    k_bn_stats<<<128, 256, 0, stream>>>(bufB, bn_sum, bn_sq, n);
    k_bn_finalize<<<1, GCN_H, 0, stream>>>(bn_sum, bn_sq, g1, beta1, bn_a, bn_c, n);
    k_bn_apply_relu_bf16<<<cdiv((long long)n * GCN_H, 256), 256, 0, stream>>>(bufB, bn_a, bn_c, bufBF, n);

    // --- layer 2: GEMM (N x 128 x 128) with bf16 A, aggregate, BN+ReLU(->bf16) ---
    k_gemm<8, true><<<gemmGrid, 256, 0, stream>>>(bufBF, W2f, bufA, n, GCN_H, GCN_H);
    k_self_init4<<<cdiv(nH4, 256), 256, 0, stream>>>((const float4*)bufA, (const float4*)b2, dinv,
                                                     (float4*)bufB, n, GCN_H / 4);
    k_edge_scatter128<<<cdiv(e, 8), 256, 0, stream>>>(src, dst, dinv, bufA, bufB, e);
    k_zero_stats<<<1, 256, 0, stream>>>(stats);
    k_bn_stats<<<128, 256, 0, stream>>>(bufB, bn_sum, bn_sq, n);
    k_bn_finalize<<<1, GCN_H, 0, stream>>>(bn_sum, bn_sq, g2, beta2, bn_a, bn_c, n);
    k_bn_apply_relu_bf16<<<cdiv((long long)n * GCN_H, 256), 256, 0, stream>>>(bufB, bn_a, bn_c, bufBF, n);

    // --- layer 3: GEMM (N x 128 x 8, zero-padded B tile), aggregate, log_softmax ---
    k_gemm<1, true><<<gemmGrid, 256, 0, stream>>>(bufBF, W3f, bufC, n, GCN_H, GCN_DOUT);
    k_self_init4<<<cdiv(nD4, 256), 256, 0, stream>>>((const float4*)bufC, (const float4*)b3, dinv,
                                                     (float4*)bufD, n, GCN_DOUT / 4);
    k_edge_scatter8<<<cdiv((long long)e * GCN_DOUT, 256), 256, 0, stream>>>(src, dst, dinv, bufC, bufD, e);
    k_log_softmax8<<<cdiv(n, 256), 256, 0, stream>>>(bufD, (float*)d_out, n);
}
